// WorldModelBackbone_43319040147923
// MI455X (gfx1250) — matrix operand order repaired and checked by reading, not compiled, not run on hardware
//
#include <hip/hip_runtime.h>
#include <cstddef>
#include <cstdint>

#define DEV __device__ __forceinline__

namespace {

constexpr int B_ = 2, T_ = 16, L_ = 256, IN_ = 768, D_ = 1024, H_ = 16, HD_ = 64;
constexpr int NREG_ = 4, P_ = NREG_ + 2, DEPTH_ = 8, TINT_ = 4, CTX_ = 30;
constexpr int HID_ = 4096, S_ = L_ + P_;
constexpr int BT_ = B_ * T_;          // 32
constexpr int MD_ = BT_ * S_;         // 8384 rows of x
constexpr float EPS_ = 1e-6f;
constexpr float BASE_ = 1000.0f;

typedef __attribute__((ext_vector_type(16))) __bf16        bfx16;
typedef __attribute__((ext_vector_type(8)))  float         fx8;
typedef __attribute__((ext_vector_type(4)))  unsigned int  ux4;
typedef __attribute__((ext_vector_type(8)))  unsigned int  ux8;

DEV unsigned short f2bf(float f) {
  unsigned u = __float_as_uint(f);
  u += 0x7fffu + ((u >> 16) & 1u);          // round-to-nearest-even
  return (unsigned short)(u >> 16);
}
DEV float silu_f(float x) { return x / (1.f + __expf(-x)); }

enum { GF_SILU = 1, GF_RESID = 2, GF_MUL = 4, GF_TOKMASK = 8, GF_OUTBF = 16, GF_BIAS = 32 };

// ---------------------------------------------------------------------------
// Issue one Tensor Data Mover load: a (tile1 x 32) bf16 tile from a row-major
// matrix (row stride = strideElems) into LDS, padded 64B-row + 16B so the LDS
// row stride is 80B (bank-conflict-free for the fragment reads).
// Must be executed by a single wave.
// ---------------------------------------------------------------------------
DEV void tdm_load_tile(const unsigned short* gsrc, unsigned lds,
                       unsigned dim0, unsigned dim1, unsigned strideElems,
                       unsigned tile1)
{
  unsigned long long ga = (unsigned long long)(const void*)gsrc;
  ux4 g0;
  g0.x = 1u;                                                 // count = 1 descriptor
  g0.y = lds;                                                // lds_addr
  g0.z = (unsigned)(ga & 0xffffffffu);                       // global_addr lo
  g0.w = (unsigned)((ga >> 32) & 0x1ffffffu) | 0x80000000u;  // addr hi | type=2
  const unsigned long long st0 = strideElems;
  ux8 g1;
  g1[0] = (1u << 16)        // data_size = 2B
        | (1u << 20)        // pad_enable
        | (3u << 22)        // pad_interval: 16 DWORDs (64B row)
        | (3u << 25);       // pad_amount: 4 DWORDs (16B) -> 80B LDS stride
  g1[1] = (dim0 & 0xffffu) << 16;                        // tensor_dim0 lo16
  g1[2] = (dim0 >> 16) | ((dim1 & 0xffffu) << 16);       // dim0 hi | tensor_dim1 lo
  g1[3] = (dim1 >> 16) | (32u << 16);                    // dim1 hi | tile_dim0 = 32
  g1[4] = tile1;                                         // tile_dim1, tile_dim2=0
  g1[5] = (unsigned)(st0 & 0xffffffffu);                 // dim0_stride lo
  g1[6] = (unsigned)((st0 >> 32) & 0xffffu)
        | ((unsigned)(st0 & 0xffffu) << 16);             // dim0_stride hi | dim1_stride lo
  g1[7] = (unsigned)((st0 >> 16) & 0xffffffffu);         // dim1_stride hi
  asm volatile("tensor_load_to_lds %0, %1" :: "s"(g0), "s"(g1) : "memory");
}

// ---------------------------------------------------------------------------
// bf16 WMMA GEMM:  C[M,N] = epilogue(A[M,K](bf16) @ B[K,N](bf16))
// B is supplied PRE-TRANSPOSED (Bt is N x K row-major), so both operand tiles
// are contiguous row-major tiles that the TDM DMAs straight into padded LDS.
// Block tile 128x256, double-buffered, TDM pipelined one k-step ahead.
// 8 wave32 waves in a 2x4 grid; each wave owns a 64x64 sub-tile (16 WMMAs/step).
// ---------------------------------------------------------------------------
__global__ __launch_bounds__(256) void k_gemm(
    const unsigned short* __restrict__ A, const unsigned short* __restrict__ Bt,
    void* __restrict__ Cout, const float* __restrict__ bias,
    const float* __restrict__ other, int M, int N, int K, int flags)
{
  constexpr int BM = 128, BN = 256, BK = 32, LST = 40;  // 40 bf16 = 80B LDS stride
  __shared__ unsigned short As[2][BM * LST];
  __shared__ unsigned short Bs[2][BN * LST];

  const int tid  = threadIdx.x;
  const int wave = tid >> 5, lane = tid & 31;
  const int lr = lane & 15, lh = lane >> 4;
  const int m0 = blockIdx.y * BM, n0 = blockIdx.x * BN;
  const int wm = wave & 1, wn = wave >> 1;              // 2x4 wave grid -> 64x64/wave

  fx8 acc[4][4];
  #pragma unroll
  for (int i = 0; i < 4; ++i)
    #pragma unroll
    for (int j = 0; j < 4; ++j) acc[i][j] = fx8{};

  const unsigned Mt = (unsigned)((M - m0) < BM ? (M - m0) : BM);
  const int nt = K / BK;

  const unsigned ldsA[2] = { (unsigned)(unsigned long long)(void*)&As[0][0],
                             (unsigned)(unsigned long long)(void*)&As[1][0] };
  const unsigned ldsB[2] = { (unsigned)(unsigned long long)(void*)&Bs[0][0],
                             (unsigned)(unsigned long long)(void*)&Bs[1][0] };

  if (wave == 0) {                                       // prologue: DMA tile 0
    tdm_load_tile(A  + (size_t)m0 * K, ldsA[0], (unsigned)K, Mt, (unsigned)K, BM);
    tdm_load_tile(Bt + (size_t)n0 * K, ldsB[0], (unsigned)K, BN, (unsigned)K, BN);
  }

  for (int t = 0; t < nt; ++t) {
    const int buf = t & 1;
    __syncthreads();                    // everyone done reading buf^1 (tile t-1)
    if (wave == 0) {
      if (t + 1 < nt) {                 // DMA tile t+1 into the other buffer
        const unsigned kt = (unsigned)(K - (t + 1) * BK);
        tdm_load_tile(A  + (size_t)m0 * K + (size_t)(t + 1) * BK, ldsA[buf ^ 1],
                      kt, Mt, (unsigned)K, BM);
        tdm_load_tile(Bt + (size_t)n0 * K + (size_t)(t + 1) * BK, ldsB[buf ^ 1],
                      kt, BN, (unsigned)K, BN);
        __builtin_amdgcn_s_wait_tensorcnt(2);   // in-order TDM: pair t has landed
      } else {
        __builtin_amdgcn_s_wait_tensorcnt(0);
      }
    }
    if (wave == 1 && lane == 0 && t + 2 < nt) {          // L2 warm-up for t+2
      __builtin_prefetch(A  + (size_t)m0 * K + (size_t)(t + 2) * BK, 0, 0);
      __builtin_prefetch(Bt + (size_t)n0 * K + (size_t)(t + 2) * BK, 0, 0);
    }
    __syncthreads();                    // tile t visible to all waves

    // fragments (ISA wave32 16-bit A 16x32 / B 32x16 layouts; 16B ds chunks)
    union Fr { bfx16 v; ux4 q[2]; };
    Fr af[4], bfr[4];
    #pragma unroll
    for (int mi = 0; mi < 4; ++mi) {
      const int m = wm * 64 + mi * 16 + lr;
      const unsigned short* rp = &As[buf][m * LST];
      af[mi].q[0] = *(const ux4*)(rp + 8 * lh);          // K = 8h..8h+7
      af[mi].q[1] = *(const ux4*)(rp + 16 + 8 * lh);     // K = 16+8h..23+8h
    }
    #pragma unroll
    for (int ni = 0; ni < 4; ++ni) {
      const int n = wn * 64 + ni * 16 + lr;
      const unsigned short* rp = &Bs[buf][n * LST + 8 * lh];  // K = 8h..8h+7, 16+8h..23+8h
      bfr[ni].q[0] = *(const ux4*)(rp);
      bfr[ni].q[1] = *(const ux4*)(rp + 16);
    }
    #pragma unroll
    for (int mi = 0; mi < 4; ++mi)
      #pragma unroll
      for (int ni = 0; ni < 4; ++ni)
        acc[mi][ni] = __builtin_amdgcn_wmma_f32_16x16x32_bf16(
            false, af[mi].v, false, bfr[ni].v, (short)0, acc[mi][ni], false, false);
  }

  // epilogue: C layout  M = e + 8*(lane/16), N = lane%16 per 16x16 tile
  #pragma unroll
  for (int mi = 0; mi < 4; ++mi) {
    #pragma unroll
    for (int ni = 0; ni < 4; ++ni) {
      const int col = n0 + wn * 64 + ni * 16 + lr;
      #pragma unroll
      for (int e = 0; e < 8; ++e) {
        const int row = m0 + wm * 64 + mi * 16 + lh * 8 + e;
        if (row < M) {
          float v = acc[mi][ni][e];
          const size_t idx = (size_t)row * N + col;
          if (flags & GF_BIAS) v += bias[col];
          if (flags & GF_SILU) v = silu_f(v);
          if (flags & GF_MUL)  v *= other[idx];
          if (flags & GF_TOKMASK) { if ((row % S_) < P_) v = 0.f; }
          if (flags & GF_RESID) v += other[idx];
          if (flags & GF_OUTBF) ((unsigned short*)Cout)[idx] = f2bf(v);
          else                  ((float*)Cout)[idx] = v;
        }
      }
    }
  }
}

// ---------------------------------------------------------------------------
// Fused convert + transpose: W[K][N] f32  ->  WT[N][K] bf16 (tiled via LDS).
// Makes the GEMM B operand a row-major tile the TDM can DMA directly.
// ---------------------------------------------------------------------------
__global__ __launch_bounds__(256) void k_wT(const float* __restrict__ W,
                                            unsigned short* __restrict__ WT,
                                            int K, int N)
{
  __shared__ float tile[32][33];
  const int n0 = blockIdx.x * 32;
  const int k0 = blockIdx.y * 32;
  const int lx = threadIdx.x & 31;
  const int ly = threadIdx.x >> 5;       // 0..7
  #pragma unroll
  for (int j = 0; j < 4; ++j)
    tile[ly * 4 + j][lx] = W[(size_t)(k0 + ly * 4 + j) * N + n0 + lx];
  __syncthreads();
  #pragma unroll
  for (int j = 0; j < 4; ++j)
    WT[(size_t)(n0 + ly * 4 + j) * K + k0 + lx] = f2bf(tile[lx][ly * 4 + j]);
}

// ---------------------------------------------------------------------------
// Build the 6 prefix rows of x per (b,t): registers, noise-MLP row, action row.
// ---------------------------------------------------------------------------
__global__ __launch_bounds__(256) void k_build_x(
    float* __restrict__ x, const float* __restrict__ nlvl, const float* __restrict__ actions,
    const unsigned char* __restrict__ amask,
    const float* __restrict__ nw1, const float* __restrict__ nb1,
    const float* __restrict__ nw2, const float* __restrict__ nb2,
    const float* __restrict__ base_ae, const float* __restrict__ aw,
    const float* __restrict__ ab, const float* __restrict__ regs)
{
  const int bt = blockIdx.x;
  const int tid = threadIdx.x;
  float* xr = x + (size_t)bt * S_ * D_;
  __shared__ float h1[D_];
  const float lvl = nlvl[bt];
  for (int j = tid; j < D_; j += 256) {
    float v = lvl * nw1[j] + nb1[j];
    h1[j] = silu_f(v);
  }
  __syncthreads();
  const bool am = amask[bt] != 0;
  for (int j = tid; j < D_; j += 256) {
    float s = nb2[j];
    for (int k = 0; k < D_; ++k) s += h1[k] * nw2[(size_t)k * D_ + j];
    xr[(size_t)NREG_ * D_ + j] = s;                         // noise row (index 4)
    float a = base_ae[j];
    if (am) {
      float aa = ab[j];
      for (int k = 0; k < 6; ++k) aa += actions[bt * 6 + k] * aw[(size_t)k * D_ + j];
      a += aa;
    }
    xr[(size_t)(NREG_ + 1) * D_ + j] = a;                   // action row (index 5)
    for (int r = 0; r < NREG_; ++r) xr[(size_t)r * D_ + j] = regs[(size_t)r * D_ + j];
  }
}

__global__ void k_f32bf(const float* __restrict__ s, unsigned short* __restrict__ d, int n) {
  for (int i = blockIdx.x * blockDim.x + threadIdx.x; i < n; i += gridDim.x * blockDim.x)
    d[i] = f2bf(s[i]);
}

__global__ void k_scatter(const float* __restrict__ src, float* __restrict__ x, int n) {
  for (int i = blockIdx.x * blockDim.x + threadIdx.x; i < n; i += gridDim.x * blockDim.x) {
    const int bt = i / (L_ * D_);
    const int rem = i - bt * (L_ * D_);
    x[(size_t)bt * S_ * D_ + (size_t)P_ * D_ + rem] = src[i];
  }
}

// RMS-norm a row of x (optionally gathering only token rows) -> bf16 operand
__global__ __launch_bounds__(256) void k_rms(
    const float* __restrict__ X, unsigned short* __restrict__ Y,
    const float* __restrict__ w, int gather)
{
  const int r = blockIdx.x;
  const float* xr;
  if (gather) { const int bt = r / L_, l = r - bt * L_; xr = X + ((size_t)bt * S_ + P_ + l) * D_; }
  else        { xr = X + (size_t)r * D_; }
  const int tid = threadIdx.x;
  float ss = 0.f;
  for (int j = tid; j < D_; j += 256) { float v = xr[j]; ss += v * v; }
  #pragma unroll
  for (int o = 16; o; o >>= 1) ss += __shfl_xor(ss, o, 32);
  __shared__ float red[8];
  if ((tid & 31) == 0) red[tid >> 5] = ss;
  __syncthreads();
  float tot = 0.f;
  #pragma unroll
  for (int i = 0; i < 8; ++i) tot += red[i];
  const float rs = rsqrtf(tot / (float)D_ + EPS_);
  unsigned short* yr = Y + (size_t)r * D_;
  for (int j = tid; j < D_; j += 256) yr[j] = f2bf(xr[j] * rs * w[j]);
}

// per-(token,head) q/k RMS-norm + RoPE (in place on qkv)
__global__ void k_qkprep(float* __restrict__ qkv, int use_tpos) {
  const int gid = blockIdx.x * blockDim.x + threadIdx.x;
  if (gid >= MD_ * H_) return;
  const int head = gid & (H_ - 1);
  const int tok  = gid >> 4;
  const int bt = tok / S_, s = tok - bt * S_;
  const int t = bt & (T_ - 1);
  const float pos = (float)(use_tpos ? t : s);
  float* base = qkv + (size_t)tok * 3 * D_ + head * HD_;
  #pragma unroll
  for (int part = 0; part < 2; ++part) {
    float* p = base + part * D_;
    float ss = 0.f;
    for (int d = 0; d < HD_; ++d) { float v = p[d]; ss += v * v; }
    const float rs = rsqrtf(ss / (float)HD_ + EPS_);
    for (int d = 0; d < HD_ / 2; ++d) {
      const float x1 = p[2 * d] * rs, x2 = p[2 * d + 1] * rs;
      const float ang = pos * __powf(BASE_, -(2.f * d) / (float)HD_);
      float sn, c;
      __sincosf(ang, &sn, &c);
      p[2 * d]     = x1 * c - x2 * sn;
      p[2 * d + 1] = x1 * sn + x2 * c;
    }
  }
}

// spatial attention: one block per (b,t,head); flash-style online softmax
__global__ __launch_bounds__(256) void k_attn_sp(const float* __restrict__ qkv,
                                                 float* __restrict__ o)
{
  const int bt = blockIdx.x >> 4;
  const int head = blockIdx.x & 15;
  const float* base = qkv + (size_t)bt * S_ * 3 * D_ + head * HD_;
  __shared__ float ks[64][HD_];
  __shared__ float vs[64][HD_];
  const int tid = threadIdx.x;
  for (int q0 = 0; q0 < S_; q0 += 256) {
    const int qi = q0 + tid;
    const bool qv = qi < S_;
    float qreg[HD_];
    if (qv) {
      const float* qp = base + (size_t)qi * 3 * D_;
      #pragma unroll
      for (int d = 0; d < HD_; ++d) qreg[d] = qp[d];
    }
    float mx = -3.0e38f, lsum = 0.f, oa[HD_];
    #pragma unroll
    for (int d = 0; d < HD_; ++d) oa[d] = 0.f;
    for (int c0 = 0; c0 < S_; c0 += 64) {
      const int cn = (S_ - c0) < 64 ? (S_ - c0) : 64;
      __syncthreads();
      for (int idx = tid; idx < cn * HD_; idx += 256) {
        const int r = idx >> 6, d = idx & 63;
        const float* kp = base + (size_t)(c0 + r) * 3 * D_;
        ks[r][d] = kp[D_ + d];
        vs[r][d] = kp[2 * D_ + d];
      }
      __syncthreads();
      if (qv) {
        for (int j = 0; j < cn; ++j) {
          const int ki = c0 + j;
          // spatial mask: all-to-all, except noise row (4) and action row (5)
          // attend only to themselves.
          const bool ok = (qi == NREG_)     ? (ki == NREG_)
                        : (qi == NREG_ + 1) ? (ki == NREG_ + 1)
                                            : true;
          if (!ok) continue;
          float sdot = 0.f;
          #pragma unroll
          for (int d = 0; d < HD_; ++d) sdot += qreg[d] * ks[j][d];
          sdot *= 0.125f;                     // 1/sqrt(64)
          const float nm = fmaxf(mx, sdot);
          const float corr = __expf(mx - nm);
          const float pw = __expf(sdot - nm);
          lsum = lsum * corr + pw;
          #pragma unroll
          for (int d = 0; d < HD_; ++d) oa[d] = oa[d] * corr + pw * vs[j][d];
          mx = nm;
        }
      }
    }
    if (qv) {
      const float inv = lsum > 0.f ? 1.f / lsum : 0.f;
      float* op = o + ((size_t)bt * S_ + qi) * D_ + head * HD_;
      #pragma unroll
      for (int d = 0; d < HD_; ++d) op[d] = oa[d] * inv;
    }
  }
}

// temporal attention: block per (b,s); thread = (head, qt); strictly-causal band
__global__ __launch_bounds__(256) void k_attn_tm(const float* __restrict__ qkv,
                                                 float* __restrict__ o,
                                                 const unsigned char* __restrict__ indep)
{
  const int b = blockIdx.x / S_;
  const int s = blockIdx.x - b * S_;
  const int head = threadIdx.x >> 4;
  const int qt = threadIdx.x & 15;
  const bool ind = indep[b] != 0;
  const float* qp = qkv + (((size_t)(b * T_ + qt) * S_ + s) * 3 * D_) + head * HD_;
  float qreg[HD_];
  #pragma unroll
  for (int d = 0; d < HD_; ++d) qreg[d] = qp[d];
  float mx = -3.0e38f, lsum = 0.f, oa[HD_];
  #pragma unroll
  for (int d = 0; d < HD_; ++d) oa[d] = 0.f;
  for (int kt = 0; kt < T_; ++kt) {
    const bool ok = (!ind) && (kt < qt) && (qt - kt <= CTX_);
    if (!ok) continue;
    const float* kp = qkv + (((size_t)(b * T_ + kt) * S_ + s) * 3 * D_) + D_ + head * HD_;
    const float* vp = kp + D_;
    float sdot = 0.f;
    #pragma unroll
    for (int d = 0; d < HD_; ++d) sdot += qreg[d] * kp[d];
    sdot *= 0.125f;
    const float nm = fmaxf(mx, sdot);
    const float corr = __expf(mx - nm);
    const float pw = __expf(sdot - nm);
    lsum = lsum * corr + pw;
    #pragma unroll
    for (int d = 0; d < HD_; ++d) oa[d] = oa[d] * corr + pw * vp[d];
    mx = nm;
  }
  const float inv = lsum > 0.f ? 1.f / lsum : 0.f;
  float* op = o + (((size_t)(b * T_ + qt) * S_ + s) * D_) + head * HD_;
  #pragma unroll
  for (int d = 0; d < HD_; ++d) op[d] = oa[d] * inv;
}

void launch_gemm(hipStream_t st, const void* A, const void* Wt, void* C,
                 const float* bias, const float* other, int M, int N, int K, int flags) {
  dim3 g(N / 256, (M + 127) / 128);
  k_gemm<<<g, 256, 0, st>>>((const unsigned short*)A, (const unsigned short*)Wt,
                            C, bias, other, M, N, K, flags);
}

void launch_wt(hipStream_t st, const float* W, unsigned short* WT, int K, int N) {
  dim3 g(N / 32, K / 32);
  k_wT<<<g, 256, 0, st>>>(W, WT, K, N);
}

} // namespace

extern "C" void kernel_launch(void* const* d_in, const int* in_sizes, int n_in,
                              void* d_out, int out_size, void* d_ws, size_t ws_size,
                              hipStream_t stream)
{
  const float* noisy   = (const float*)d_in[0];
  const float* nlvl    = (const float*)d_in[1];
  const float* actions = (const float*)d_in[2];
  const unsigned char* indep = (const unsigned char*)d_in[3];
  const unsigned char* amask = (const unsigned char*)d_in[4];
  const float* W_in   = (const float*)d_in[5];
  const float* nw1    = (const float*)d_in[6];
  const float* nb1    = (const float*)d_in[7];
  const float* nw2    = (const float*)d_in[8];
  const float* nb2    = (const float*)d_in[9];
  const float* base_ae = (const float*)d_in[10];
  const float* aw     = (const float*)d_in[11];
  const float* ab     = (const float*)d_in[12];
  const float* regs   = (const float*)d_in[13];
  const float* ln1    = (const float*)d_in[14];
  const float* Wqkv_s = (const float*)d_in[15];
  const float* Wo_s   = (const float*)d_in[16];
  const float* lnt    = (const float*)d_in[17];
  const float* Wqkv_t = (const float*)d_in[18];
  const float* Wo_t   = (const float*)d_in[19];
  const float* ln2    = (const float*)d_in[20];
  const float* W1     = (const float*)d_in[21];
  const float* W2     = (const float*)d_in[22];
  const float* W3     = (const float*)d_in[23];
  const float* finw   = (const float*)d_in[24];
  const float* W_out  = (const float*)d_in[25];
  const float* b_out  = (const float*)d_in[26];
  (void)in_sizes; (void)n_in; (void)out_size; (void)ws_size;

  char* wsb = (char*)d_ws;
  size_t off = 0;
  auto alloc = [&](size_t bytes) -> void* {
    void* p = wsb + off;
    off += (bytes + 255) & ~(size_t)255;
    return p;
  };
  float*          x    = (float*)alloc((size_t)MD_ * D_ * 4);        // activations
  float*          qkv  = (float*)alloc((size_t)MD_ * 3 * D_ * 4);
  float*          obuf = (float*)alloc((size_t)MD_ * D_ * 4);        // attn out / embed tmp
  float*          t1   = (float*)alloc((size_t)MD_ * HID_ * 4);      // silu(h@W1)
  unsigned short* Abf  = (unsigned short*)alloc((size_t)MD_ * D_ * 2);
  unsigned short* Ab2  = (unsigned short*)alloc((size_t)MD_ * HID_ * 2);
  unsigned short* wbf  = (unsigned short*)alloc((size_t)D_ * HID_ * 2);  // Wt (N x K bf16)

  // --- embedding assembly ---
  k_build_x<<<BT_, 256, 0, stream>>>(x, nlvl, actions, amask, nw1, nb1, nw2, nb2,
                                     base_ae, aw, ab, regs);
  k_f32bf<<<2048, 256, 0, stream>>>(noisy, Abf, BT_ * L_ * IN_);
  launch_wt(stream, W_in, wbf, IN_, D_);
  launch_gemm(stream, Abf, wbf, obuf, nullptr, nullptr, BT_ * L_, D_, IN_, 0);
  k_scatter<<<2048, 256, 0, stream>>>(obuf, x, BT_ * L_ * D_);

  for (int i = 0; i < DEPTH_; ++i) {
    // --- spatial attention ---
    k_rms<<<MD_, 256, 0, stream>>>(x, Abf, ln1 + (size_t)i * D_, 0);
    launch_wt(stream, Wqkv_s + (size_t)i * D_ * 3 * D_, wbf, D_, 3 * D_);
    launch_gemm(stream, Abf, wbf, qkv, nullptr, nullptr, MD_, 3 * D_, D_, 0);
    k_qkprep<<<(MD_ * H_ + 255) / 256, 256, 0, stream>>>(qkv, 0);
    k_attn_sp<<<BT_ * H_, 256, 0, stream>>>(qkv, obuf);
    k_f32bf<<<2048, 256, 0, stream>>>(obuf, Abf, MD_ * D_);
    launch_wt(stream, Wo_s + (size_t)i * D_ * D_, wbf, D_, D_);
    launch_gemm(stream, Abf, wbf, x, nullptr, x, MD_, D_, D_, GF_RESID);

    // --- temporal attention every TINT layers ---
    if (i % TINT_ == 0) {
      k_rms<<<MD_, 256, 0, stream>>>(x, Abf, lnt + (size_t)i * D_, 0);
      launch_wt(stream, Wqkv_t + (size_t)i * D_ * 3 * D_, wbf, D_, 3 * D_);
      launch_gemm(stream, Abf, wbf, qkv, nullptr, nullptr, MD_, 3 * D_, D_, 0);
      k_qkprep<<<(MD_ * H_ + 255) / 256, 256, 0, stream>>>(qkv, 1);
      k_attn_tm<<<B_ * S_, 256, 0, stream>>>(qkv, obuf, indep);
      k_f32bf<<<2048, 256, 0, stream>>>(obuf, Abf, MD_ * D_);
      launch_wt(stream, Wo_t + (size_t)i * D_ * D_, wbf, D_, D_);
      launch_gemm(stream, Abf, wbf, x, nullptr, x, MD_, D_, D_, GF_RESID | GF_TOKMASK);
    }

    // --- gated MLP ---
    k_rms<<<MD_, 256, 0, stream>>>(x, Abf, ln2 + (size_t)i * D_, 0);
    launch_wt(stream, W1 + (size_t)i * D_ * HID_, wbf, D_, HID_);
    launch_gemm(stream, Abf, wbf, t1, nullptr, nullptr, MD_, HID_, D_, GF_SILU);
    launch_wt(stream, W2 + (size_t)i * D_ * HID_, wbf, D_, HID_);
    launch_gemm(stream, Abf, wbf, Ab2, nullptr, t1, MD_, HID_, D_, GF_MUL | GF_OUTBF);
    launch_wt(stream, W3 + (size_t)i * HID_ * D_, wbf, HID_, D_);
    launch_gemm(stream, Ab2, wbf, x, nullptr, x, MD_, D_, HID_, GF_RESID);
  }

  // --- final norm over token rows + output projection ---
  k_rms<<<BT_ * L_, 256, 0, stream>>>(x, Abf, finw, 1);
  launch_wt(stream, W_out, wbf, D_, IN_);
  launch_gemm(stream, Abf, wbf, d_out, b_out, nullptr, BT_ * L_, IN_, D_, GF_BIAS);
}
